// Affine_Module_Abla_22067541967130
// MI455X (gfx1250) — matrix operand order repaired and checked by reading, not compiled, not run on hardware
//
#include <hip/hip_runtime.h>
#include <hip/hip_bf16.h>

typedef __attribute__((ext_vector_type(16))) __bf16 v16bf;
typedef __attribute__((ext_vector_type(8)))  __bf16 v8bf;
typedef __attribute__((ext_vector_type(8)))  float  v8f;
typedef __attribute__((ext_vector_type(4)))  float  f4;
typedef __attribute__((ext_vector_type(2)))  float  f2;

#define NB    8
#define NPTS  4096
#define DIN   128
#define DOUT  256
#define LDA_BF 136   /* bf16 a-tile stride: DIN+8  (272B rows, 16B aligned, bank-rotated) */
#define LDY_BF 264   /* bf16 y-tile stride: DOUT+8 (528B rows, 16B aligned, bank-rotated) */

// LDS layout (dynamic):
//   [0, 25344)          : lds_ybf  48 x LDY_BF bf16  (Y in bf16, GEMM2 A-operand)
//   [25344, 25344+49152): union { lds_abf 48 x LDA_BF bf16 (phase 0/1),
//                                 lds_out 256*3*16 f32      (phase 2+) }
#define LDS_YBF_BYTES (48 * LDY_BF * 2)
#define LDS_U_BYTES   (DOUT * 3 * 16 * 4)
#define LDS_TOTAL     (LDS_YBF_BYTES + LDS_U_BYTES)

__device__ __forceinline__ v8f wmma_bf16(v16bf a, v16bf b, v8f c) {
  // D(f32 16x16) = A(16x32 bf16) * B(32x16 bf16) + C
  return __builtin_amdgcn_wmma_f32_16x16x32_bf16(
      /*neg_a=*/false, a, /*neg_b=*/false, b,
      /*c_mod=*/(short)0, c, /*reuse_a=*/false, /*reuse_b=*/false);
}

// A-fragment (16x32 bf16) straight from a bf16 LDS tile: two ds_load_b128.
// Contiguous-K order, matching load_bfrag: lane m=lane%16, g=lane/16; elems e <-> K=k0+16g+e.
__device__ __forceinline__ v16bf load_afrag_bf(const __bf16* lds, int ld, int row,
                                               int k0, int g) {
  const __bf16* p = lds + row * ld + k0 + 16 * g;
  v8bf lo = *(const v8bf*)(p);
  v8bf hi = *(const v8bf*)(p + 8);
  v16bf r;
  #pragma unroll
  for (int e = 0; e < 8; ++e) { r[e] = lo[e]; r[e + 8] = hi[e]; }
  return r;
}

// B-fragment (32x16 bf16): lane n=lane%16, g=lane/16; elems e <-> K=k0+16g+e.
// Row-major weight [n][k] == K-contiguous column => one 32B contiguous load.
__device__ __forceinline__ v16bf load_bfrag(const __bf16* w, int ld, int ncol,
                                            int k0, int g) {
  return *(const v16bf*)(w + (size_t)ncol * ld + k0 + 16 * g);
}

// ---------------- weight prep: M = bf16(A+B+C), Wb = bf16(Wdir) -------------
__global__ __launch_bounds__(256) void prep_weights(
    const float* __restrict__ A, const float* __restrict__ B,
    const float* __restrict__ C, const float* __restrict__ W,
    __bf16* __restrict__ Mb, __bf16* __restrict__ Wb) {
  int i = blockIdx.x * 256 + threadIdx.x;
  if (i < DOUT * DIN)  Mb[i] = (__bf16)(A[i] + B[i] + C[i]);
  if (i < DOUT * DOUT) Wb[i] = (__bf16)W[i];
}

// ---------------- fused rotation + GEMM1 + GEMM2 + VNLeakyReLU --------------
__global__ __launch_bounds__(256) void fused_vn(
    const float* __restrict__ X, const float* __restrict__ J,
    const __bf16* __restrict__ Mb, const __bf16* __restrict__ Wb,
    float* __restrict__ out) {
  extern __shared__ char smem[];
  __bf16* lds_ybf = (__bf16*)smem;
  __bf16* lds_abf = (__bf16*)(smem + LDS_YBF_BYTES);
  float*  lds_out = (float*)(smem + LDS_YBF_BYTES);  // aliases lds_abf (after barrier)

  const int tid = threadIdx.x;
  const int wg  = blockIdx.x;
  const int b   = wg >> 8;            // NPTS/16 = 256 tiles per batch
  const int n0  = (wg & 255) << 4;

  // ---- phase 0: rotation + a = R^T x, converted once to bf16 in lds_abf --
  // rows: r = comp*16 + point ; cols: d (Din)
  #pragma unroll
  for (int k = 0; k < 8; ++k) {
    const int task = tid + k * 256;   // 16 points * 128 d = 2048 tasks
    const int p = task >> 7;
    const int d = task & 127;
    const size_t base = (size_t)(b * NPTS + n0 + p) * DIN + d;
    const float* xp = X + base * 3;
    const f2*    jp = (const f2*)(J + base * 6);  // 24B record, 8B aligned
    const float x0 = xp[0], x1 = xp[1], x2 = xp[2];
    const f2 j01 = jp[0], j23 = jp[1], j45 = jp[2];
    const float a10 = j01.x, a20 = j01.y;
    const float a11 = j23.x, a21 = j23.y;
    const float a12 = j45.x, a22 = j45.y;
    // b1 = normalize(col0)
    float nrm = sqrtf(a10 * a10 + a11 * a11 + a12 * a12);
    float inv = 1.0f / fmaxf(nrm, 1e-12f);
    const float b10 = a10 * inv, b11 = a11 * inv, b12 = a12 * inv;
    // b2 = normalize(col1 - (b1.col1) b1)
    const float pr = b10 * a20 + b11 * a21 + b12 * a22;
    const float u0 = a20 - pr * b10, u1 = a21 - pr * b11, u2 = a22 - pr * b12;
    nrm = sqrtf(u0 * u0 + u1 * u1 + u2 * u2);
    inv = 1.0f / fmaxf(nrm, 1e-12f);
    const float b20 = u0 * inv, b21 = u1 * inv, b22 = u2 * inv;
    // b3 = b1 x b2
    const float b30 = b11 * b22 - b12 * b21;
    const float b31 = b12 * b20 - b10 * b22;
    const float b32 = b10 * b21 - b11 * b20;
    lds_abf[(0 * 16 + p) * LDA_BF + d] = (__bf16)(b10 * x0 + b11 * x1 + b12 * x2);
    lds_abf[(1 * 16 + p) * LDA_BF + d] = (__bf16)(b20 * x0 + b21 * x1 + b22 * x2);
    lds_abf[(2 * 16 + p) * LDA_BF + d] = (__bf16)(b30 * x0 + b31 * x1 + b32 * x2);
  }
  __syncthreads();

  const int wave = tid >> 5;
  const int lane = tid & 31;
  const int g    = lane >> 4;
  const int hn   = lane & 15;

  // f32 Y accumulators persist in registers: the wave that computes Y[:,ct]
  // is the same wave that needs x = Y[:,ct] in the elementwise phase.
  v8f xacc[2][3];

  // ---- phase 1: Y[48 x 256] = a[48 x 128] * M^T  (bf16 WMMA, f32 acc) ----
  #pragma unroll
  for (int ci = 0; ci < 2; ++ci) {
    const int ct = wave + (ci << 3);          // col-tile (16 out channels)
    v8f y0 = {0, 0, 0, 0, 0, 0, 0, 0};
    v8f y1 = y0, y2 = y0;
    #pragma unroll
    for (int ks = 0; ks < 4; ++ks) {
      const int k0 = ks << 5;
      const v16bf bf = load_bfrag(Mb, DIN, ct * 16 + hn, k0, g);
      const v16bf a0 = load_afrag_bf(lds_abf, LDA_BF,  0 + hn, k0, g);
      const v16bf a1 = load_afrag_bf(lds_abf, LDA_BF, 16 + hn, k0, g);
      const v16bf a2 = load_afrag_bf(lds_abf, LDA_BF, 32 + hn, k0, g);
      y0 = wmma_bf16(a0, bf, y0);
      y1 = wmma_bf16(a1, bf, y1);
      y2 = wmma_bf16(a2, bf, y2);
    }
    xacc[ci][0] = y0; xacc[ci][1] = y1; xacc[ci][2] = y2;
    // bf16 copy of Y for GEMM2 A-fragments (each element converted exactly once)
    // C layout: VGPR v, lane -> M = v + 8*(lane/16), N = lane%16
    #pragma unroll
    for (int v = 0; v < 8; ++v) {
      const int m   = v + (g << 3);
      const int col = ct * 16 + hn;
      lds_ybf[( 0 + m) * LDY_BF + col] = (__bf16)y0[v];
      lds_ybf[(16 + m) * LDY_BF + col] = (__bf16)y1[v];
      lds_ybf[(32 + m) * LDY_BF + col] = (__bf16)y2[v];
    }
  }
  __syncthreads();

  // ---- phase 2: d = Y * W^T ; VNLeakyReLU (x from registers) -------------
  #pragma unroll
  for (int ci = 0; ci < 2; ++ci) {
    const int ct = wave + (ci << 3);
    v8f d0 = {0, 0, 0, 0, 0, 0, 0, 0};
    v8f d1 = d0, d2 = d0;
    #pragma unroll 4
    for (int ks = 0; ks < 8; ++ks) {
      const int k0 = ks << 5;
      const v16bf bf = load_bfrag(Wb, DOUT, ct * 16 + hn, k0, g);
      const v16bf a0 = load_afrag_bf(lds_ybf, LDY_BF,  0 + hn, k0, g);
      const v16bf a1 = load_afrag_bf(lds_ybf, LDY_BF, 16 + hn, k0, g);
      const v16bf a2 = load_afrag_bf(lds_ybf, LDY_BF, 32 + hn, k0, g);
      d0 = wmma_bf16(a0, bf, d0);
      d1 = wmma_bf16(a1, bf, d1);
      d2 = wmma_bf16(a2, bf, d2);
    }
    #pragma unroll
    for (int v = 0; v < 8; ++v) {
      const int m  = v + (g << 3);
      const int ol = ct * 16 + hn;                 // output channel o
      const float xv0 = xacc[ci][0][v];
      const float xv1 = xacc[ci][1][v];
      const float xv2 = xacc[ci][2][v];
      const float dv0 = d0[v], dv1 = d1[v], dv2 = d2[v];
      const float dot  = xv0 * dv0 + xv1 * dv1 + xv2 * dv2;
      const float dnsq = dv0 * dv0 + dv1 * dv1 + dv2 * dv2;
      const float corr = dot / (dnsq + 1e-6f);
      const bool  pos  = (dot >= 0.0f);
      const float z0 = pos ? xv0 : fmaf(-corr, dv0, xv0);
      const float z1 = pos ? xv1 : fmaf(-corr, dv1, xv1);
      const float z2 = pos ? xv2 : fmaf(-corr, dv2, xv2);
      lds_out[(ol * 3 + 0) * 16 + m] = 0.2f * xv0 + 0.8f * z0;
      lds_out[(ol * 3 + 1) * 16 + m] = 0.2f * xv1 + 0.8f * z1;
      lds_out[(ol * 3 + 2) * 16 + m] = 0.2f * xv2 + 0.8f * z2;
    }
  }
  __syncthreads();

  // ---- coalesced writeback: out[b, o, i, n0..n0+15] (64B runs) -----------
  for (int q = tid; q < DOUT * 3; q += 256) {
    const int tt = q >> 8;          // component i
    const int o  = q & 255;         // output channel
    const f4* s = (const f4*)(lds_out + (o * 3 + tt) * 16);
    f4* dg = (f4*)(out + ((size_t)(b * DOUT + o) * 3 + tt) * (size_t)NPTS + n0);
    dg[0] = s[0];
    dg[1] = s[1];
    dg[2] = s[2];
    dg[3] = s[3];
  }
}

extern "C" void kernel_launch(void* const* d_in, const int* in_sizes, int n_in,
                              void* d_out, int out_size, void* d_ws, size_t ws_size,
                              hipStream_t stream) {
  const float* X  = (const float*)d_in[0];
  const float* J  = (const float*)d_in[1];
  const float* Am = (const float*)d_in[2];
  const float* Bm = (const float*)d_in[3];
  const float* Cm = (const float*)d_in[4];
  const float* W  = (const float*)d_in[5];
  float* out = (float*)d_out;

  __bf16* Mb = (__bf16*)d_ws;                                   // 256x128 bf16
  __bf16* Wb = (__bf16*)((char*)d_ws + (size_t)DOUT * DIN * 2); // 256x256 bf16

  prep_weights<<<(DOUT * DOUT + 255) / 256, 256, 0, stream>>>(Am, Bm, Cm, W, Mb, Wb);

  const int nblocks = NB * NPTS / 16;  // 2048 workgroups, 16 points each
  fused_vn<<<nblocks, 256, LDS_TOTAL, stream>>>(X, J, Mb, Wb, out);
}
